// MultiHeadAttention_35476429865965
// MI455X (gfx1250) — compile-verified, matrix operand order checked by reading, not run on hardware
//
#include <hip/hip_runtime.h>
#include <hip/hip_bf16.h>

// ---------------- problem constants ----------------
#define BATCH 4
#define SEQ   2048
#define DIM   1024
#define NHEAD 16
#define HDIM  64
#define MTOT  (BATCH * SEQ)        // 8192

typedef __bf16 bf16;
typedef __attribute__((ext_vector_type(16))) bf16  v16bf;
typedef __attribute__((ext_vector_type(8)))  bf16  v8bf;
typedef __attribute__((ext_vector_type(8)))  float v8f;
typedef __attribute__((ext_vector_type(4)))  int   v4i;

#define AS1 __attribute__((address_space(1)))
#define AS3 __attribute__((address_space(3)))

typedef AS1 v4i* gv4i_ptr;   // pointer to v4i in global (AS1)
typedef AS3 v4i* lv4i_ptr;   // pointer to v4i in LDS (AS3)

// ---- CDNA5 async global->LDS copy (ASYNCcnt path), guarded ----
#if defined(__has_builtin)
#if __has_builtin(__builtin_amdgcn_global_load_async_to_lds_b128)
#define HAVE_ASYNC_LDS 1
#endif
#endif

__device__ __forceinline__ void stage_b128(const bf16* gsrc, bf16* ldst) {
#ifdef HAVE_ASYNC_LDS
    __builtin_amdgcn_global_load_async_to_lds_b128(
        (gv4i_ptr)(uintptr_t)gsrc,
        (lv4i_ptr)(unsigned int)(uintptr_t)ldst, 0, 0);
#else
    *(v8bf*)ldst = *(const v8bf*)gsrc;   // sync fallback: b128 load + ds store
#endif
}

__device__ __forceinline__ void async_wait0() {
#ifdef HAVE_ASYNC_LDS
#if __has_builtin(__builtin_amdgcn_s_wait_asynccnt)
    __builtin_amdgcn_s_wait_asynccnt(0);
#else
    asm volatile("s_wait_asynccnt 0x0" ::: "memory");
#endif
#endif
}

// float -> bf16, round-to-nearest-even
__device__ __forceinline__ bf16 f2bf(float f) {
    union { float f; unsigned u; } in; in.f = f;
    unsigned u = in.u;
    u += 0x7FFFu + ((u >> 16) & 1u);
    union { unsigned short s; bf16 b; } out;
    out.s = (unsigned short)(u >> 16);
    return out.b;
}

// Load one 16x32 bf16 WMMA fragment (A layout; also valid as B when memory
// holds the B matrix transposed, i.e. memory row n = B column n) from a
// row-major [16 x >=32] region with leading dimension `ld` (elements).
// ISA 16-bit layout: lanes 0-15 k=0..7/16..23, lanes 16-31 k=8..15/24..31.
__device__ __forceinline__ v16bf load_frag(const bf16* p, int ld, int lane) {
    int row = lane & 15;
    int kb  = (lane >> 4) << 3;           // 0 or 8
    v8bf lo = *reinterpret_cast<const v8bf*>(p + row * ld + kb);
    v8bf hi = *reinterpret_cast<const v8bf*>(p + row * ld + kb + 16);
    v16bf r;
#pragma unroll
    for (int i = 0; i < 8; ++i) { r[i] = lo[i]; r[i + 8] = hi[i]; }
    return r;
}

__device__ __forceinline__ v8f wmma_bf16(v16bf a, v16bf b, v8f c) {
    return __builtin_amdgcn_wmma_f32_16x16x32_bf16(
        false, a, false, b, (short)0, c, false, false);
}

// ---------------- kernel 1: f32 -> bf16 convert ----------------
__global__ void k_cvt_bf16(const float* __restrict__ in, bf16* __restrict__ out, int n) {
    int i = blockIdx.x * blockDim.x + threadIdx.x;
    int stride = gridDim.x * blockDim.x;
    for (; i < n; i += stride) out[i] = f2bf(in[i]);
}

// ---------------- kernel 2: projection GEMM  C = X * W^T + b ----------------
// Block = 8 waves sharing one 64-wide W strip (n0..n0+63), each wave owns one
// 16-row M tile. W strip k-slices (64x32 = 4KB) are double-buffered in LDS via
// async global->LDS copies (one b128 per thread per step).
// MODE 0: store bf16 as [B, H, S, Hd]   (Q and K)
// MODE 1: store bf16 as [B, H, Hd, S]   (V transposed, for the P*V GEMM)
// MODE 2: store f32  as [MTOT, DIM]     (final output projection)
template <int MODE>
__global__ __launch_bounds__(256, 2)
void k_proj(const bf16* __restrict__ X, const bf16* __restrict__ W,
            const float* __restrict__ bias, void* __restrict__ outv) {
    __shared__ bf16 wbuf[2][64][32];      // 8 KB, double-buffered W strip

    int tid  = threadIdx.x;
    int lane = tid & 31;
    int w    = tid >> 5;                  // wave in block
    int ns   = blockIdx.x & 15;           // n strip
    int mg   = blockIdx.x >> 4;           // m group (8 tiles)
    int n0   = ns * 64;                   // aligned to a head
    int m0   = (mg * 8 + w) * 16;

    // staging map: thread t -> W row n0 + (t>>2), k chunk (t&3)*8
    int snr = tid >> 2;
    int skc = (tid & 3) * 8;
    const bf16* wsrc = W + (size_t)(n0 + snr) * DIM + skc;

    v8f acc0 = {}, acc1 = {}, acc2 = {}, acc3 = {};
    const bf16* xp = X + (size_t)m0 * DIM;

    stage_b128(wsrc, &wbuf[0][snr][skc]);
    async_wait0();
    __syncthreads();

    int cur = 0;
    for (int k0 = 0; k0 < DIM; k0 += 32) {
        if (k0 + 32 < DIM)
            stage_b128(wsrc + k0 + 32, &wbuf[cur ^ 1][snr][skc]);

        v16bf a = load_frag(xp + k0, DIM, lane);
        acc0 = wmma_bf16(a, load_frag(&wbuf[cur][ 0][0], 32, lane), acc0);
        acc1 = wmma_bf16(a, load_frag(&wbuf[cur][16][0], 32, lane), acc1);
        acc2 = wmma_bf16(a, load_frag(&wbuf[cur][32][0], 32, lane), acc2);
        acc3 = wmma_bf16(a, load_frag(&wbuf[cur][48][0], 32, lane), acc3);

        async_wait0();
        __syncthreads();
        cur ^= 1;
    }

    // C layout: lane L -> column n = L&15 ; VGPR r -> row r + (L>=16 ? 8 : 0)
    int nc = lane & 15;
    int rowbase = (lane >> 4) * 8;
    v8f accs[4] = {acc0, acc1, acc2, acc3};
#pragma unroll
    for (int t = 0; t < 4; ++t) {
        int n = n0 + t * 16 + nc;
        int h = n >> 6, d = n & 63;
        float bv = bias[n];
#pragma unroll
        for (int r = 0; r < 8; ++r) {
            int mg2 = m0 + rowbase + r;
            int b   = mg2 >> 11;                 // / SEQ
            int s   = mg2 & (SEQ - 1);
            float val = accs[t][r] + bv;
            if (MODE == 0)
                ((bf16*)outv)[(((size_t)(b * NHEAD + h)) * SEQ + s) * HDIM + d] = f2bf(val);
            else if (MODE == 1)
                ((bf16*)outv)[(((size_t)(b * NHEAD + h)) * HDIM + d) * SEQ + s] = f2bf(val);
            else
                ((float*)outv)[(size_t)mg2 * DIM + n] = val;
        }
    }
}

// ---------------- kernel 3: flash attention ----------------
// Block = 8 waves sharing one (b,h); each wave owns one 16-query tile.
// K (32x64) and V^T (64x32) kv-slices are double-buffered in LDS via async
// copies; each wave runs online softmax + two WMMA GEMM chains per slice.
__global__ __launch_bounds__(256, 2)
void k_attn(const bf16* __restrict__ Q, const bf16* __restrict__ K,
            const bf16* __restrict__ Vt, bf16* __restrict__ att) {
    __shared__ bf16 kbuf[2][32][64];      // 8 KB
    __shared__ bf16 vbuf[2][64][32];      // 8 KB
    __shared__ bf16 pbuf[8][16][32];      // 8 KB, per-wave P transpose staging

    int tid  = threadIdx.x;
    int lane = tid & 31;
    int wib  = tid >> 5;
    int bh   = blockIdx.x >> 4;           // 0..63  (= b*NHEAD + h)
    int qg   = blockIdx.x & 15;           // query-tile group
    int qt   = qg * 8 + wib;              // this wave's 16-row query tile

    const bf16* Qp = Q  + ((size_t)bh * SEQ + qt * 16) * HDIM;
    const bf16* Kp = K  + (size_t)bh * SEQ * HDIM;
    const bf16* Vp = Vt + (size_t)bh * HDIM * SEQ;    // [HDIM, SEQ]

    // staging maps (one b128 of K and one of V per thread per slice)
    int kr = tid >> 3, kc = (tid & 7) * 8;            // K: 32 rows x 64
    int vr = tid >> 2, sc = (tid & 3) * 8;            // V: 64 rows x 32

    v16bf a0 = load_frag(Qp,      HDIM, lane);        // q k = 0..31
    v16bf a1 = load_frag(Qp + 32, HDIM, lane);        // q k = 32..63

    v8f o0 = {}, o1 = {}, o2 = {}, o3 = {};
    float mrow[8], lrow[8];
#pragma unroll
    for (int r = 0; r < 8; ++r) { mrow[r] = -1e30f; lrow[r] = 0.0f; }

    const float kscale = 0.125f * 1.44269504088896340736f;  // 1/sqrt(Hd)*log2(e)

    stage_b128(Kp + (size_t)kr * HDIM + kc, &kbuf[0][kr][kc]);
    stage_b128(Vp + (size_t)vr * SEQ + sc,  &vbuf[0][vr][sc]);
    async_wait0();
    __syncthreads();

    int cur = 0;
    for (int j0 = 0; j0 < SEQ; j0 += 32) {
        if (j0 + 32 < SEQ) {
            stage_b128(Kp + (size_t)(j0 + 32 + kr) * HDIM + kc, &kbuf[cur ^ 1][kr][kc]);
            stage_b128(Vp + (size_t)vr * SEQ + (j0 + 32) + sc,  &vbuf[cur ^ 1][vr][sc]);
        }

        // scores: two 16x16 tiles over kv columns [j0, j0+16) and [j0+16, j0+32)
        v8f s0 = {}, s1 = {};
        s0 = wmma_bf16(a0, load_frag(&kbuf[cur][ 0][ 0], 64, lane), s0);
        s0 = wmma_bf16(a1, load_frag(&kbuf[cur][ 0][32], 64, lane), s0);
        s1 = wmma_bf16(a0, load_frag(&kbuf[cur][16][ 0], 64, lane), s1);
        s1 = wmma_bf16(a1, load_frag(&kbuf[cur][16][32], 64, lane), s1);

        float alpha[8];
#pragma unroll
        for (int r = 0; r < 8; ++r) {
            float x0 = s0[r] * kscale;
            float x1 = s1[r] * kscale;
            float mx = fmaxf(x0, x1);
            mx = fmaxf(mx, __shfl_xor(mx, 1, 16));
            mx = fmaxf(mx, __shfl_xor(mx, 2, 16));
            mx = fmaxf(mx, __shfl_xor(mx, 4, 16));
            mx = fmaxf(mx, __shfl_xor(mx, 8, 16));
            float mnew = fmaxf(mrow[r], mx);
            float a  = exp2f(mrow[r] - mnew);
            float p0 = exp2f(x0 - mnew);
            float p1 = exp2f(x1 - mnew);
            float rs = p0 + p1;
            rs += __shfl_xor(rs, 1, 16);
            rs += __shfl_xor(rs, 2, 16);
            rs += __shfl_xor(rs, 4, 16);
            rs += __shfl_xor(rs, 8, 16);
            lrow[r] = lrow[r] * a + rs;
            mrow[r] = mnew;
            alpha[r] = a;
            s0[r] = p0;   // reuse score regs to hold P
            s1[r] = p1;
        }
#pragma unroll
        for (int r = 0; r < 8; ++r) {
            o0[r] *= alpha[r]; o1[r] *= alpha[r];
            o2[r] *= alpha[r]; o3[r] *= alpha[r];
        }

        // transpose P (C layout -> A fragment layout) through per-wave LDS
        int nc2 = lane & 15;
        int rowbase = (lane >> 4) * 8;
#pragma unroll
        for (int r = 0; r < 8; ++r) {
            pbuf[wib][rowbase + r][nc2]      = f2bf(s0[r]);
            pbuf[wib][rowbase + r][16 + nc2] = f2bf(s1[r]);
        }
        v16bf ap;   // same-wave RAW through LDS: DS ops in-order per wave
        {
            int row = lane & 15;
            int kb  = (lane >> 4) << 3;
            v8bf lo = *reinterpret_cast<const v8bf*>(&pbuf[wib][row][kb]);
            v8bf hi = *reinterpret_cast<const v8bf*>(&pbuf[wib][row][16 + kb]);
#pragma unroll
            for (int i = 0; i < 8; ++i) { ap[i] = lo[i]; ap[i + 8] = hi[i]; }
        }

        // O += P(16x32) * V(32x64)  (V^T rows are d -> row-major fragments)
        o0 = wmma_bf16(ap, load_frag(&vbuf[cur][ 0][0], 32, lane), o0);
        o1 = wmma_bf16(ap, load_frag(&vbuf[cur][16][0], 32, lane), o1);
        o2 = wmma_bf16(ap, load_frag(&vbuf[cur][32][0], 32, lane), o2);
        o3 = wmma_bf16(ap, load_frag(&vbuf[cur][48][0], 32, lane), o3);

        async_wait0();
        __syncthreads();
        cur ^= 1;
    }

    // epilogue: normalize and store to att[B, S, D] (heads re-interleaved)
    int b = bh >> 4;
    int h = bh & 15;
    int nc = lane & 15;
    int rowbase = (lane >> 4) * 8;
#pragma unroll
    for (int r = 0; r < 8; ++r) {
        float inv = 1.0f / lrow[r];
        int s = qt * 16 + rowbase + r;
        size_t base = ((size_t)(b * SEQ + s)) * DIM + h * 64;
        att[base +  0 + nc] = f2bf(o0[r] * inv);
        att[base + 16 + nc] = f2bf(o1[r] * inv);
        att[base + 32 + nc] = f2bf(o2[r] * inv);
        att[base + 48 + nc] = f2bf(o3[r] * inv);
    }
}

// ---------------- host side ----------------
extern "C" void kernel_launch(void* const* d_in, const int* in_sizes, int n_in,
                              void* d_out, int out_size, void* d_ws, size_t ws_size,
                              hipStream_t stream) {
    (void)in_sizes; (void)n_in; (void)out_size; (void)ws_size;

    const float* x  = (const float*)d_in[0];
    const float* Wq = (const float*)d_in[1];
    const float* bq = (const float*)d_in[2];
    const float* Wk = (const float*)d_in[3];
    const float* bk = (const float*)d_in[4];
    const float* Wv = (const float*)d_in[5];
    const float* bv = (const float*)d_in[6];
    const float* Wo = (const float*)d_in[7];
    const float* bo = (const float*)d_in[8];

    const size_t MD = (size_t)MTOT * DIM;   // 8388608 elems
    const size_t DD = (size_t)DIM * DIM;    // 1048576 elems

    bf16* Xb  = (bf16*)d_ws;
    bf16* Wqb = Xb  + MD;
    bf16* Wkb = Wqb + DD;
    bf16* Wvb = Wkb + DD;
    bf16* Wob = Wvb + DD;
    bf16* Qb  = Wob + DD;
    bf16* Kb  = Qb  + MD;
    bf16* Vtb = Kb  + MD;
    bf16* Att = Vtb + MD;
    // total: (5*MD + 4*DD) * 2B ~= 92 MB of workspace

    // 1) precision conversion (fp32 inputs -> bf16 WMMA operands)
    k_cvt_bf16<<<2048, 256, 0, stream>>>(x,  Xb,  (int)MD);
    k_cvt_bf16<<<512,  256, 0, stream>>>(Wq, Wqb, (int)DD);
    k_cvt_bf16<<<512,  256, 0, stream>>>(Wk, Wkb, (int)DD);
    k_cvt_bf16<<<512,  256, 0, stream>>>(Wv, Wvb, (int)DD);
    k_cvt_bf16<<<512,  256, 0, stream>>>(Wo, Wob, (int)DD);

    // 2) QKV projections (1024 blocks: 64 m-groups x 16 n-strips)
    k_proj<0><<<1024, 256, 0, stream>>>(Xb, Wqb, bq, Qb);
    k_proj<0><<<1024, 256, 0, stream>>>(Xb, Wkb, bk, Kb);
    k_proj<1><<<1024, 256, 0, stream>>>(Xb, Wvb, bv, Vtb);

    // 3) flash attention (1024 blocks: 64 (b,h) x 16 query-tile groups)
    k_attn<<<1024, 256, 0, stream>>>(Qb, Kb, Vtb, Att);

    // 4) output projection -> f32 d_out
    k_proj<2><<<1024, 256, 0, stream>>>(Att, Wob, bo, d_out);
}
